// PointMLPEncoderAdapter_77429670412908
// MI455X (gfx1250) — compile-verified, matrix operand order checked by reading, not run on hardware
//
#include <hip/hip_runtime.h>
#include <math.h>

// ---------------------------------------------------------------------------
// Types for CDNA5 WMMA
// ---------------------------------------------------------------------------
typedef __attribute__((ext_vector_type(16))) __bf16    v16bf;
typedef __attribute__((ext_vector_type(8)))  float     v8f;
typedef __attribute__((ext_vector_type(8)))  unsigned  v8u;

__device__ __forceinline__ __bf16 f2bf(float f) {
  unsigned u = __builtin_bit_cast(unsigned, f);
  unsigned r = u + 0x7FFFu + ((u >> 16) & 1u);   // round-to-nearest-even
  unsigned short h = (unsigned short)(r >> 16);
  return __builtin_bit_cast(__bf16, h);
}
__device__ __forceinline__ unsigned pk(float lo, float hi) {
  unsigned a = (unsigned)__builtin_bit_cast(unsigned short, f2bf(lo));
  unsigned b = (unsigned)__builtin_bit_cast(unsigned short, f2bf(hi));
  return a | (b << 16);
}

// ---------------------------------------------------------------------------
// Generic GEMM: Y[g,o,n] = sum_c W[o,c] * X[g,c,n]  (bf16 WMMA, fp32 accum)
// Requirements: C % 32 == 0, O % 64 == 0, Ns % 16 == 0 (Ns = padded N stride).
// One wave computes a 64(M) x 16(N) tile: 4 WMMAs per 32-deep K chunk.
// Optional per-output-channel sum / sumsq accumulation (BN), masked to n < Np.
// ---------------------------------------------------------------------------
__global__ __launch_bounds__(32)
void gemm_wmma(const float* __restrict__ W, const float* __restrict__ X,
               float* __restrict__ Y, float* __restrict__ sum,
               float* __restrict__ sqs, int O, int C, int Np, int Ns) {
  const int g    = blockIdx.z;
  const int o0   = blockIdx.y << 6;
  const int n0   = blockIdx.x << 4;
  const int lane = threadIdx.x;
  const int m    = lane & 15;
  const int hi   = lane >> 4;

  __shared__ unsigned sAw[64 * 16];   // A tile: 64 rows x 32 K, bf16 pairs
  __shared__ unsigned sBw[16 * 16];   // B tile (transposed): 16 n x 32 K
  __bf16* sBb = reinterpret_cast<__bf16*>(sBw);

  const float* Xg = X + (size_t)g * C * Ns;
  v8f acc[4] = {v8f{}, v8f{}, v8f{}, v8f{}};

  for (int kk = 0; kk < C; kk += 32) {
    // ---- fill A tile (W), convert to bf16 on the way in -------------------
#pragma unroll
    for (int q = 0; q < 16; ++q) {
      int idx = lane + 32 * q;                 // 0..511 float4s
      int r   = idx >> 3;                      // row 0..63
      int c4  = idx & 7;                       // float4 within row
      float4 wv = *(const float4*)(W + (size_t)(o0 + r) * C + kk + c4 * 4);
      sAw[r * 16 + c4 * 2]     = pk(wv.x, wv.y);
      sAw[r * 16 + c4 * 2 + 1] = pk(wv.z, wv.w);
    }
    // ---- fill B tile (X), transposed to n-major ---------------------------
#pragma unroll
    for (int q = 0; q < 4; ++q) {
      int idx = lane + 32 * q;                 // 0..127 float4s
      int k   = idx >> 2;                      // K row 0..31
      int seg = idx & 3;                       // n group of 4
      float4 xv = *(const float4*)(Xg + (size_t)(kk + k) * Ns + n0 + seg * 4);
      sBb[(seg * 4 + 0) * 32 + k] = f2bf(xv.x);
      sBb[(seg * 4 + 1) * 32 + k] = f2bf(xv.y);
      sBb[(seg * 4 + 2) * 32 + k] = f2bf(xv.z);
      sBb[(seg * 4 + 3) * 32 + k] = f2bf(xv.w);
    }
    __syncthreads();

    // B fragment (shared by the 4 M-tiles)
    v8u bu;
#pragma unroll
    for (int v = 0; v < 8; ++v) {
      int kw = (v < 4) ? (hi * 4 + v) : (4 + hi * 4 + v);
      bu[v] = sBw[m * 16 + kw];
    }
    v16bf bfrag = __builtin_bit_cast(v16bf, bu);

#pragma unroll
    for (int mt = 0; mt < 4; ++mt) {
      v8u au;
#pragma unroll
      for (int v = 0; v < 8; ++v) {
        int kw = (v < 4) ? (hi * 4 + v) : (4 + hi * 4 + v);
        au[v] = sAw[(mt * 16 + m) * 16 + kw];
      }
      v16bf afrag = __builtin_bit_cast(v16bf, au);
      acc[mt] = __builtin_amdgcn_wmma_f32_16x16x32_bf16(
          false, afrag, false, bfrag, (short)0, acc[mt], false, false);
    }
    __syncthreads();
  }

  // ---- epilogue: store + (masked) BN statistics ---------------------------
  float* Yg = Y + (size_t)g * O * Ns + n0 + m;
#pragma unroll
  for (int mt = 0; mt < 4; ++mt) {
#pragma unroll
    for (int v = 0; v < 8; ++v) {
      int mm = o0 + mt * 16 + v + hi * 8;
      Yg[(size_t)mm * Ns] = acc[mt][v];
    }
  }
  if (sum != nullptr && (n0 + m) < Np) {
#pragma unroll
    for (int mt = 0; mt < 4; ++mt) {
#pragma unroll
      for (int v = 0; v < 8; ++v) {
        int mm = o0 + mt * 16 + v + hi * 8;
        float val = acc[mt][v];
        atomicAdd(&sum[mm], val);
        atomicAdd(&sqs[mm], val * val);
      }
    }
  }
}

// ---------------------------------------------------------------------------
// Small helper kernels
// ---------------------------------------------------------------------------
__global__ void zero_f(float* p, long n) {
  long i = (long)blockIdx.x * 256 + threadIdx.x;
  if (i < n) p[i] = 0.f;
}

__global__ void bn_finalize(float* sum, float* sqs, int O, float invcnt) {
  int o = blockIdx.x * 64 + threadIdx.x;
  if (o >= O) return;
  float mean = sum[o] * invcnt;
  float var  = sqs[o] * invcnt - mean * mean;
  sum[o] = mean;
  sqs[o] = rsqrtf(fmaxf(var, 0.f) + 1e-5f);
}

// Elementwise BN apply (+optional residual, +optional ReLU); Nl = layout stride.
__global__ void bn_apply(const float* __restrict__ src, float* __restrict__ dst,
                         const float* __restrict__ gamma,
                         const float* __restrict__ beta,
                         const float* __restrict__ mean,
                         const float* __restrict__ inv,
                         const float* __restrict__ res,
                         int relu, long total, int O, int Nl) {
  long i = (long)blockIdx.x * 256 + threadIdx.x;
  if (i >= total) return;
  int o = (int)((i / Nl) % O);
  float v = gamma[o] * (src[i] - mean[o]) * inv[o] + beta[o];
  if (res) v += res[i];
  if (relu) v = fmaxf(v, 0.f);
  dst[i] = v;
}

// point_cloud (B,1024,3) -> padded (B,32,1024), channels 3..31 zero.
__global__ void transpose_pc_pad(const float* __restrict__ pc,
                                 float* __restrict__ X) {
  long i = (long)blockIdx.x * 256 + threadIdx.x;   // i = (b*32+c)*1024+n
  if (i >= 32L * 32 * 1024) return;
  int n = (int)(i % 1024);
  int c = (int)((i / 1024) % 32);
  int b = (int)(i / (32 * 1024));
  X[i] = (c < 3) ? pc[((long)b * 1024 + n) * 3 + c] : 0.f;
}

// embedding weight (64,3) -> padded (64,32)
__global__ void pad_w(const float* __restrict__ W, float* __restrict__ Wp) {
  int i = blockIdx.x * 256 + threadIdx.x;          // i = r*32+c
  if (i >= 64 * 32) return;
  int c = i & 31, r = i >> 5;
  Wp[i] = (c < 3) ? W[r * 3 + c] : 0.f;
}

// Farthest point sampling: one block per batch element.
__global__ void fps_kernel(const float* __restrict__ xyz, int Nn, int np,
                           int* __restrict__ out) {
  int b = blockIdx.x;
  __shared__ float dist[1024];
  __shared__ float rv[256];
  __shared__ int   ri[256];
  __shared__ int   far;
  for (int n = threadIdx.x; n < Nn; n += 256) dist[n] = 1e10f;
  if (threadIdx.x == 0) far = 0;
  __syncthreads();
  for (int it = 0; it < np; ++it) {
    int f = far;
    if (threadIdx.x == 0) out[b * np + it] = f;
    const float* ct = xyz + ((long)b * Nn + f) * 3;
    float cx = ct[0], cy = ct[1], cz = ct[2];
    float lm = -1.f; int li = 0;
    for (int n = threadIdx.x; n < Nn; n += 256) {
      const float* p = xyz + ((long)b * Nn + n) * 3;
      float dx = p[0] - cx, dy = p[1] - cy, dz = p[2] - cz;
      float d  = dx * dx + dy * dy + dz * dz;
      float nd = fminf(dist[n], d);
      dist[n] = nd;
      if (nd > lm) { lm = nd; li = n; }
    }
    rv[threadIdx.x] = lm; ri[threadIdx.x] = li;
    __syncthreads();
    for (int off = 128; off > 0; off >>= 1) {
      if (threadIdx.x < off) {
        float ov = rv[threadIdx.x + off];
        int   oi = ri[threadIdx.x + off];
        if (ov > rv[threadIdx.x] ||
            (ov == rv[threadIdx.x] && oi < ri[threadIdx.x])) {
          rv[threadIdx.x] = ov; ri[threadIdx.x] = oi;
        }
      }
      __syncthreads();
    }
    if (threadIdx.x == 0) far = ri[0];
    __syncthreads();
  }
}

__global__ void gather_xyz(const float* __restrict__ xyz,
                           const int* __restrict__ idx,
                           float* __restrict__ nxyz, int Nn, int S) {
  int t = blockIdx.x * 256 + threadIdx.x;
  if (t >= 32 * S) return;
  int b = t / S;
  int nid = idx[t];
  const float* p = xyz + ((long)b * Nn + nid) * 3;
  nxyz[(long)t * 3 + 0] = p[0];
  nxyz[(long)t * 3 + 1] = p[1];
  nxyz[(long)t * 3 + 2] = p[2];
}

__global__ void gather_np(const float* __restrict__ feats,
                          const int* __restrict__ idx,
                          float* __restrict__ NP, int C, int Nn, int S) {
  long i = (long)blockIdx.x * 256 + threadIdx.x;
  if (i >= 32L * S * C) return;
  int c = (int)(i % C);
  int s = (int)((i / C) % S);
  int b = (int)(i / ((long)C * S));
  int nid = idx[b * S + s];
  NP[i] = feats[((long)b * C + c) * Nn + nid];
}

// k=24 nearest neighbours via per-anchor insertion sort.
__global__ void knn_kernel(const float* __restrict__ xyz,
                           const float* __restrict__ nxyz,
                           int* __restrict__ knn, int Nn, int S) {
  int t = blockIdx.x * 64 + threadIdx.x;
  if (t >= 32 * S) return;
  int b = t / S;
  float qx = nxyz[(long)t * 3 + 0];
  float qy = nxyz[(long)t * 3 + 1];
  float qz = nxyz[(long)t * 3 + 2];
  float bd[24]; int bi[24];
  for (int j = 0; j < 24; ++j) { bd[j] = 1e30f; bi[j] = 0; }
  for (int n = 0; n < Nn; ++n) {
    const float* p = xyz + ((long)b * Nn + n) * 3;
    float dx = p[0] - qx, dy = p[1] - qy, dz = p[2] - qz;
    float d  = dx * dx + dy * dy + dz * dz;
    if (d < bd[23]) {
      int j = 23;
      while (j > 0 && bd[j - 1] > d) {
        bd[j] = bd[j - 1]; bi[j] = bi[j - 1]; --j;
      }
      bd[j] = d; bi[j] = n;
    }
  }
  for (int j = 0; j < 24; ++j) knn[(long)t * 24 + j] = bi[j];
}

// Build centered grouped features (+replicate half) with K-stride 32 padding;
// reduce per-batch sum/sumsq of the centered values.
__global__ void group_center(const float* __restrict__ feats,
                             const float* __restrict__ NP,
                             const int* __restrict__ knn,
                             float* __restrict__ X, float* bsum, float* bsqs,
                             int C, int Nn, int S) {
  long i = (long)blockIdx.x * 256 + threadIdx.x;
  int  k  = (int)(i % 24);
  long r  = i / 24;
  int  c  = (int)(r % C);
  long r2 = r / C;
  int  s  = (int)(r2 % S);
  int  b  = (int)(r2 / S);
  long g  = (long)b * S + s;
  float np  = NP[g * C + c];
  int   nid = knn[g * 24 + k];
  float ctr = feats[((long)b * C + c) * Nn + nid] - np;
  X[(g * 2 * C + c) * 32 + k]     = ctr;
  X[(g * 2 * C + C + c) * 32 + k] = np;
  __shared__ float s1[256], s2[256];
  s1[threadIdx.x] = ctr;
  s2[threadIdx.x] = ctr * ctr;
  __syncthreads();
  for (int off = 128; off > 0; off >>= 1) {
    if (threadIdx.x < off) {
      s1[threadIdx.x] += s1[threadIdx.x + off];
      s2[threadIdx.x] += s2[threadIdx.x + off];
    }
    __syncthreads();
  }
  if (threadIdx.x == 0) {   // all threads of a block share the same b
    atomicAdd(&bsum[b], s1[0]);
    atomicAdd(&bsqs[b], s2[0]);
  }
}

__global__ void batch_std_finalize(float* bsum, float* bsqs, long cnt) {
  int b = threadIdx.x;
  if (b >= 32) return;
  float m   = bsum[b] / (float)cnt;
  float var = (bsqs[b] - (float)cnt * m * m) / (float)(cnt - 1);
  bsum[b] = 1.f / (sqrtf(fmaxf(var, 0.f)) + 1e-5f);
}

__global__ void group_scale(float* __restrict__ X,
                            const float* __restrict__ alpha,
                            const float* __restrict__ beta,
                            const float* __restrict__ binv, int C, int S) {
  long i = (long)blockIdx.x * 256 + threadIdx.x;
  if (i >= 32L * S * C * 24) return;
  int  k  = (int)(i % 24);
  long r  = i / 24;
  int  c  = (int)(r % C);
  long r2 = r / C;
  int  s  = (int)(r2 % S);
  int  b  = (int)(r2 / S);
  long g  = (long)b * S + s;
  float* px = &X[(g * 2 * C + c) * 32 + k];
  *px = alpha[c] * (*px) * binv[b] + beta[c];
}

__global__ void maxpool_k(const float* __restrict__ T, float* __restrict__ F,
                          int O, int S) {
  long i = (long)blockIdx.x * 256 + threadIdx.x;
  if (i >= 32L * O * S) return;
  int s = (int)(i % S);
  int o = (int)((i / S) % O);
  int b = (int)(i / ((long)S * O));
  const float* t = T + (((long)b * S + s) * O + o) * 32;
  float m = t[0];
  for (int k = 1; k < 24; ++k) m = fmaxf(m, t[k]);
  F[((long)b * O + o) * S + s] = m;
}

// 3-NN interpolation of the (already projected) 256-ch features + BN stats.
__global__ __launch_bounds__(32)
void interp_kernel(const float* __restrict__ txyz, const float* __restrict__ sxyz,
                   const float* __restrict__ P, float* __restrict__ preOut,
                   float* sum, float* sqs) {
  int bn = blockIdx.x;
  int b = bn >> 10, n = bn & 1023;
  __shared__ int   sid[3];
  __shared__ float sw[3];
  if (threadIdx.x == 0) {
    const float* q = txyz + ((long)b * 1024 + n) * 3;
    float qx = q[0], qy = q[1], qz = q[2];
    float d0 = 1e30f, d1 = 1e30f, d2 = 1e30f;
    int   i0 = 0, i1 = 0, i2 = 0;
    for (int s = 0; s < 64; ++s) {
      const float* p = sxyz + ((long)b * 64 + s) * 3;
      float dx = p[0] - qx, dy = p[1] - qy, dz = p[2] - qz;
      float d  = dx * dx + dy * dy + dz * dz;
      if (d < d0)      { d2 = d1; i2 = i1; d1 = d0; i1 = i0; d0 = d; i0 = s; }
      else if (d < d1) { d2 = d1; i2 = i1; d1 = d; i1 = s; }
      else if (d < d2) { d2 = d; i2 = s; }
    }
    float w0 = 1.f / (d0 + 1e-8f), w1 = 1.f / (d1 + 1e-8f), w2 = 1.f / (d2 + 1e-8f);
    float wsm = w0 + w1 + w2;
    sid[0] = i0; sid[1] = i1; sid[2] = i2;
    sw[0] = w0 / wsm; sw[1] = w1 / wsm; sw[2] = w2 / wsm;
  }
  __syncthreads();
  for (int o = threadIdx.x; o < 256; o += 32) {
    const float* Pb = P + ((long)b * 256 + o) * 64;
    float v = sw[0] * Pb[sid[0]] + sw[1] * Pb[sid[1]] + sw[2] * Pb[sid[2]];
    preOut[((long)b * 256 + o) * 1024 + n] = v;
    atomicAdd(&sum[o], v);
    atomicAdd(&sqs[o], v * v);
  }
}

__global__ void final_store(const float* __restrict__ pre,
                            const float* __restrict__ gamma,
                            const float* __restrict__ beta,
                            const float* __restrict__ mean,
                            const float* __restrict__ inv,
                            float* __restrict__ out) {
  long i = (long)blockIdx.x * 256 + threadIdx.x;
  if (i >= 32L * 256 * 1024) return;
  int n = (int)(i % 1024);
  int o = (int)((i / 1024) % 256);
  int b = (int)(i / (1024L * 256));
  float v  = gamma[o] * (pre[i] - mean[o]) * inv[o] + beta[o];
  float ge = 0.5f * v * (1.f + erff(v * 0.70710678118f));
  out[((long)b * 1024 + n) * 256 + o] = ge;
}

// ---------------------------------------------------------------------------
// Host orchestration
// ---------------------------------------------------------------------------
extern "C" void kernel_launch(void* const* d_in, const int* in_sizes, int n_in,
                              void* d_out, int out_size, void* d_ws, size_t ws_size,
                              hipStream_t stream) {
  (void)in_sizes; (void)n_in; (void)out_size; (void)ws_size;

  const int SS[4]   = {512, 256, 128, 64};
  const int CIN[4]  = {64, 128, 256, 512};
  const int COUT[4] = {128, 256, 512, 1024};

  const float* PC   = (const float*)d_in[0];
  const float* embW = (const float*)d_in[1];
  const float* embB = (const float*)d_in[2];
  const float* embG = (const float*)d_in[3];
  const float* opW  = (const float*)d_in[4];
  const float* opB  = (const float*)d_in[5];
  const float* opG  = (const float*)d_in[6];
  // per-stage leaves (jax pytree, sorted dict keys): alpha,beta, pos[0..1]{c1,c2}{W,b,g},
  // pre.blocks[0..1]{c1,c2}{W,b,g}, pre.transfer{W,b,g}  -> 29 leaves / stage.
  auto SP = [&](int s, int j) { return (const float*)d_in[7 + s * 29 + j]; };

  // workspace carve-up (floats)
  float* wsf = (float*)d_ws;
  size_t off = 0;
  auto alloc = [&](size_t nf) { float* p = wsf + off; off += nf; return p; };
  const size_t BIG = 67108864;            // 16384 groups * 128 ch * 32 (padded k)
  float* bufA   = alloc(BIG);
  float* bufB   = alloc(BIG);
  float* bufC   = alloc(BIG);
  float* ftA    = alloc(2097152);
  float* ftB    = alloc(2097152);
  float* NP     = alloc(1048576);
  float* xyzA   = alloc(98304);
  float* xyzB   = alloc(98304);
  float* stats  = alloc(2048);            // [0..1023]=sum, [1024..2047]=sumsq
  float* bstats = alloc(64);              // [0..31]=sum, [32..63]=sumsq
  float* Pbuf   = alloc(524288);
  float* preOut = alloc(8388608);
  float* wpad   = alloc(2048);            // padded embedding weight (64x32)
  int*   fpsIdx = (int*)alloc(16384);
  int*   knnIdx = (int*)alloc(393216);

  auto zero = [&](float* p, long n) {
    zero_f<<<(unsigned)((n + 255) / 256), 256, 0, stream>>>(p, n);
  };
  auto run_gemm = [&](const float* W, const float* X, float* Y, int G, int O,
                      int C, int Np, int Ns, bool st) {
    if (st) zero(stats, 2048);
    dim3 grid(Ns / 16, O / 64, G);
    gemm_wmma<<<grid, 32, 0, stream>>>(W, X, Y, st ? stats : nullptr,
                                       st ? stats + 1024 : nullptr, O, C, Np, Ns);
  };
  auto run_bn = [&](int O, long cnt) {
    bn_finalize<<<(O + 63) / 64, 64, 0, stream>>>(stats, stats + 1024, O,
                                                  1.f / (float)cnt);
  };
  auto run_apply = [&](const float* src, float* dst, const float* g,
                       const float* bb, const float* res, int relu, long G,
                       int O, int Ns) {
    long tot = G * O * Ns;
    bn_apply<<<(unsigned)((tot + 255) / 256), 256, 0, stream>>>(
        src, dst, g, bb, stats, stats + 1024, res, relu, tot, O, Ns);
  };
  auto conv_bn_relu = [&](const float* W, const float* bb, const float* g,
                          const float* X, float* Y, int G, int O, int C,
                          int Np, int Ns) {
    run_gemm(W, X, Y, G, O, C, Np, Ns, true);
    run_bn(O, (long)G * Np);
    run_apply(Y, Y, g, bb, nullptr, 1, G, O, Ns);
  };
  auto res_block = [&](const float* x, float* tmp, float* t2, float* dst,
                       const float* W1, const float* b1, const float* g1,
                       const float* W2, const float* b2, const float* g2,
                       int G, int O, int Np, int Ns) {
    conv_bn_relu(W1, b1, g1, x, tmp, G, O, O, Np, Ns);
    run_gemm(W2, tmp, t2, G, O, O, Np, Ns, true);
    run_bn(O, (long)G * Np);
    run_apply(t2, dst, g2, b2, x, 1, G, O, Ns);   // relu(bn(conv2) + x)
  };

  // ---- embedding: padded (B,32,1024) -> feats (B,64,1024) ----
  pad_w<<<(64 * 32 + 255) / 256, 256, 0, stream>>>(embW, wpad);
  transpose_pc_pad<<<(32 * 32 * 1024 + 255) / 256, 256, 0, stream>>>(PC, bufC);
  conv_bn_relu(wpad, embB, embG, bufC, ftA, 32, 64, 32, 1024, 1024);

  const float* xyz_cur = PC;
  int Nin = 1024;
  float* fcur = ftA;

  for (int s = 0; s < 4; ++s) {
    const int S_ = SS[s], Cin = CIN[s], Cout = COUT[s];
    const int G = 32 * S_;
    float* xyz_new = (s % 2 == 0) ? xyzA : xyzB;

    // FPS + gathers + KNN
    fps_kernel<<<32, 256, 0, stream>>>(xyz_cur, Nin, S_, fpsIdx);
    gather_xyz<<<(32 * S_ + 255) / 256, 256, 0, stream>>>(xyz_cur, fpsIdx,
                                                          xyz_new, Nin, S_);
    gather_np<<<(unsigned)((32L * S_ * Cin + 255) / 256), 256, 0, stream>>>(
        fcur, fpsIdx, NP, Cin, Nin, S_);
    knn_kernel<<<(32 * S_ + 63) / 64, 64, 0, stream>>>(xyz_cur, xyz_new,
                                                       knnIdx, Nin, S_);

    // grouping: center, per-batch std, affine normalize, concat rep -> bufA
    zero(bstats, 64);
    long totg = 32L * S_ * Cin * 24;
    group_center<<<(unsigned)(totg / 256), 256, 0, stream>>>(
        fcur, NP, knnIdx, bufA, bstats, bstats + 32, Cin, Nin, S_);
    batch_std_finalize<<<1, 32, 0, stream>>>(bstats, bstats + 32,
                                             (long)S_ * 24 * Cin);
    group_scale<<<(unsigned)((totg + 255) / 256), 256, 0, stream>>>(
        bufA, SP(s, 0), SP(s, 1), bstats, Cin, S_);

    // pre-extraction: transfer conv then two res blocks over (G, Cout, 24|32)
    conv_bn_relu(SP(s, 26), SP(s, 27), SP(s, 28), bufA, bufB, G, Cout,
                 2 * Cin, 24, 32);
    res_block(bufB, bufC, bufA, bufA, SP(s, 14), SP(s, 15), SP(s, 16),
              SP(s, 17), SP(s, 18), SP(s, 19), G, Cout, 24, 32);
    res_block(bufA, bufC, bufB, bufB, SP(s, 20), SP(s, 21), SP(s, 22),
              SP(s, 23), SP(s, 24), SP(s, 25), G, Cout, 24, 32);

    // max over k=24 -> feats (B, Cout, S)
    float* fnew = (fcur == ftA) ? ftB : ftA;
    maxpool_k<<<(unsigned)((32L * Cout * S_ + 255) / 256), 256, 0, stream>>>(
        bufB, fnew, Cout, S_);

    // pos res blocks over (B, Cout, S)
    res_block(fnew, bufA, bufC, fcur, SP(s, 2), SP(s, 3), SP(s, 4), SP(s, 5),
              SP(s, 6), SP(s, 7), 32, Cout, S_, S_);
    res_block(fcur, bufA, bufC, fnew, SP(s, 8), SP(s, 9), SP(s, 10), SP(s, 11),
              SP(s, 12), SP(s, 13), 32, Cout, S_, S_);

    fcur = fnew;
    xyz_cur = xyz_new;
    Nin = S_;
  }

  // ---- out_proj then 3-NN interpolation (projection is linear, so commute) ----
  run_gemm(opW, fcur, Pbuf, 32, 256, 1024, 64, 64, false);   // P = Wout @ feats
  zero(stats, 2048);
  interp_kernel<<<32 * 1024, 32, 0, stream>>>(PC, xyz_cur, Pbuf, preOut, stats,
                                              stats + 1024);
  run_bn(256, 32L * 1024);
  final_store<<<(unsigned)((32L * 256 * 1024 + 255) / 256), 256, 0, stream>>>(
      preOut, opG, opB, stats, stats + 1024, (float*)d_out);
}
